// NSAAttention_82781199663132
// MI455X (gfx1250) — matrix operand order repaired and checked by reading, not compiled, not run on hardware
//
#include <hip/hip_runtime.h>

#define DEV_INLINE __device__ __forceinline__

typedef __bf16 bf16_t;
typedef __attribute__((ext_vector_type(16))) __bf16 v16bf;
typedef __attribute__((ext_vector_type(8))) float v8f;
typedef __attribute__((ext_vector_type(4))) unsigned int uint4v;
typedef __attribute__((ext_vector_type(4))) int v4i;

constexpr int S = 1024, DIM = 1024, H = 16, G = 4, HPG = 4, DK = 64, DV = 64;
constexpr int L_CMP = 32, D_STR = 16, L_SEL = 64, N_SEL = 8, W_WIN = 512;
constexpr int S_CMP = 63, N_BLK = 16, GATE_H = 32;
constexpr float SCALE = 0.125f;
constexpr int KV = G * DK; // 256

union Frag { v16bf v; unsigned int u[8]; };

DEV_INLINE unsigned int pk2(float lo, float hi) {
  union { bf16_t h[2]; unsigned int u; } x;
  x.h[0] = (bf16_t)lo; x.h[1] = (bf16_t)hi; return x.u;
}
DEV_INLINE unsigned int pk2b(bf16_t lo, bf16_t hi) {
  union { bf16_t h[2]; unsigned int u; } x;
  x.h[0] = lo; x.h[1] = hi; return x.u;
}

// -------- CDNA5 async global->LDS path (guarded; sync LDS staging fallback)
#if __has_builtin(__builtin_amdgcn_global_load_async_to_lds_b128) && \
    __has_builtin(__builtin_amdgcn_s_wait_asynccnt)
#define USE_ASYNC_LDS 1
#else
#define USE_ASYNC_LDS 0
#endif

DEV_INLINE void cp16_to_lds(const bf16_t* gp, bf16_t* lp) {
#if USE_ASYNC_LDS
  __builtin_amdgcn_global_load_async_to_lds_b128((v4i*)(void*)(gp), (v4i*)(void*)(lp), 0, 0);
#else
  *(uint4v*)lp = *(const uint4v*)gp;
#endif
}
DEV_INLINE void wait_async() {
#if USE_ASYNC_LDS
  __builtin_amdgcn_s_wait_asynccnt(0);
#endif
}

// ---------------------------------------------------------------- casts
__global__ __launch_bounds__(256) void k_cvt_bf16(const float* __restrict__ src,
                                                  bf16_t* __restrict__ dst, int n) {
  int i = blockIdx.x * 256 + threadIdx.x;
  if (i < n) dst[i] = (bf16_t)src[i];
}

// -------------------------------------------- tiled WMMA GEMM  C = A * W^T
// A: [M,K] bf16 row-major, W: [N,K] bf16 row-major, C: [M,N] f32 row-major.
// Block: 256 threads (8 waves), macro tile 128(M) x 64(N), K-slab 32,
// double-buffered LDS fed by async global->LDS copies. Wave tile: 32x32.
constexpr int BM = 128, BN = 64, BK = 32;

__global__ __launch_bounds__(256) void k_gemm_bf16(const bf16_t* __restrict__ A,
                                                   const bf16_t* __restrict__ W,
                                                   float* __restrict__ C,
                                                   int M, int N, int K) {
  __shared__ bf16_t As[2][BM][BK];
  __shared__ bf16_t Bs[2][BN][BK];
  int tid = threadIdx.x;
  int lane = tid & 31, wv = tid >> 5;
  int hi = lane >> 4, ln = lane & 15;
  int wm = wv >> 1, wn = wv & 1; // 4 x 2 wave grid
  int nBlk = N / BN;
  int bm = blockIdx.x / nBlk, bn = blockIdx.x - bm * nBlk;

  auto stage = [&](int buf, int k0) {
#pragma unroll
    for (int c = 0; c < 2; ++c) { // A: 128x32 -> 512 chunks of 16B
      int idx = tid + c * 256;
      int r = idx >> 2, q = idx & 3;
      cp16_to_lds(A + (size_t)(bm * BM + r) * K + k0 + q * 8, &As[buf][r][q * 8]);
    }
    { // B: 64x32 -> 256 chunks of 16B
      int r = tid >> 2, q = tid & 3;
      cp16_to_lds(W + (size_t)(bn * BN + r) * K + k0 + q * 8, &Bs[buf][r][q * 8]);
    }
  };

  v8f acc[2][2] = {};
  int nk = K / BK;
  stage(0, 0);
  wait_async();
  __syncthreads();
  for (int kb = 0; kb < nk; ++kb) {
    int buf = kb & 1;
    if (kb + 1 < nk) stage(buf ^ 1, (kb + 1) * BK);
    Frag a0, a1, b0, b1;
#pragma unroll
    for (int j = 0; j < 8; ++j) {
      int dd = ((j < 4) ? 2 * j : 16 + 2 * (j - 4)) + (hi ? 8 : 0);
      a0.u[j] = *(const unsigned int*)&As[buf][wm * 32 + ln][dd];
      a1.u[j] = *(const unsigned int*)&As[buf][wm * 32 + 16 + ln][dd];
    }
#pragma unroll
    for (int j = 0; j < 8; ++j) {
      int dd = ((j < 4) ? 2 * j : 16 + 2 * (j - 4)) + (hi ? 8 : 0);
      b0.u[j] = *(const unsigned int*)&Bs[buf][wn * 32 + ln][dd];
      b1.u[j] = *(const unsigned int*)&Bs[buf][wn * 32 + 16 + ln][dd];
    }
    acc[0][0] = __builtin_amdgcn_wmma_f32_16x16x32_bf16(false, a0.v, false, b0.v, (short)0, acc[0][0], false, false);
    acc[0][1] = __builtin_amdgcn_wmma_f32_16x16x32_bf16(false, a0.v, false, b1.v, (short)0, acc[0][1], false, false);
    acc[1][0] = __builtin_amdgcn_wmma_f32_16x16x32_bf16(false, a1.v, false, b0.v, (short)0, acc[1][0], false, false);
    acc[1][1] = __builtin_amdgcn_wmma_f32_16x16x32_bf16(false, a1.v, false, b1.v, (short)0, acc[1][1], false, false);
    wait_async();
    __syncthreads();
  }
#pragma unroll
  for (int mi = 0; mi < 2; ++mi)
#pragma unroll
    for (int t = 0; t < 2; ++t) {
      int nn = bn * BN + wn * 32 + t * 16 + ln;
#pragma unroll
      for (int r = 0; r < 8; ++r) {
        int mm = bm * BM + wm * 32 + mi * 16 + r + hi * 8;
        C[(size_t)mm * N + nn] = acc[mi][t][r];
      }
    }
}

// ---------------------------------------------------------------- RoPE
__global__ __launch_bounds__(256) void k_rope_q(float* __restrict__ q,
                                                bf16_t* __restrict__ qbf) {
  int i = blockIdx.x * 256 + threadIdx.x;
  if (i >= S * H * 32) return;
  int d = i & 31, h = (i >> 5) & (H - 1), s = i >> 9;
  float inv = __powf(10000.0f, -(float)d * (1.0f / 32.0f));
  float ang = (float)s * inv;
  float sn, c;
  __sincosf(ang, &sn, &c);
  float* p = q + ((size_t)s * H + h) * DK;
  float x1 = p[d], x2 = p[d + 32];
  float y1 = x1 * c - x2 * sn;
  float y2 = x1 * sn + x2 * c;
  p[d] = y1; p[d + 32] = y2;
  bf16_t* pb = qbf + ((size_t)s * H + h) * DK;
  pb[d] = (bf16_t)y1; pb[d + 32] = (bf16_t)y2;
}

__global__ __launch_bounds__(256) void k_rope_k(float* __restrict__ k,
                                                bf16_t* __restrict__ kbf) {
  int i = blockIdx.x * 256 + threadIdx.x;
  if (i >= S * G * 32) return;
  int d = i & 31, g = (i >> 5) & (G - 1), s = i >> 7;
  float inv = __powf(10000.0f, -(float)d * (1.0f / 32.0f));
  float ang = (float)s * inv;
  float sn, c;
  __sincosf(ang, &sn, &c);
  float* p = k + (size_t)s * KV + g * DK;
  float x1 = p[d], x2 = p[d + 32];
  float y1 = x1 * c - x2 * sn;
  float y2 = x1 * sn + x2 * c;
  p[d] = y1; p[d + 32] = y2;
  bf16_t* pb = kbf + (size_t)s * KV + g * DK;
  pb[d] = (bf16_t)y1; pb[d + 32] = (bf16_t)y2;
}

// ------------------------------------------------------------- mean pool
__global__ __launch_bounds__(256) void k_pool(const float* __restrict__ kc,
                                              const float* __restrict__ vc,
                                              bf16_t* __restrict__ Kcb,
                                              bf16_t* __restrict__ Vcb) {
  int i = blockIdx.x * 256 + threadIdx.x;
  if (i >= G * 64 * 64) return;
  int d = i & 63, c = (i >> 6) & 63, g = i >> 12;
  float ka = 0.f, va = 0.f;
  if (c < S_CMP) {
    int base = c * D_STR;
    for (int t = 0; t < L_CMP; ++t) {
      ka += kc[(size_t)(base + t) * KV + g * DK + d];
      va += vc[(size_t)(base + t) * KV + g * DK + d];
    }
    ka *= (1.0f / L_CMP);
    va *= (1.0f / L_CMP);
  }
  Kcb[i] = (bf16_t)ka;
  Vcb[i] = (bf16_t)va;
}

// ------------------------------------------- compressed attention + p_slc
__global__ __launch_bounds__(128) void k_cmp_attn(const bf16_t* __restrict__ qbf,
                                                  const bf16_t* __restrict__ Kcb,
                                                  const bf16_t* __restrict__ Vcb,
                                                  float* __restrict__ Ocmp,
                                                  float* __restrict__ pslc) {
  __shared__ float lp[4][16][64];
  int w = threadIdx.x >> 5, lane = threadIdx.x & 31;
  int hi = lane >> 4, ln = lane & 15;
  int wid = blockIdx.x * 4 + w;
  int g = wid & (G - 1), sT = wid >> 2;
  float pacc[8];
#pragma unroll
  for (int q8 = 0; q8 < 8; ++q8) pacc[q8] = 0.f;

  for (int hp = 0; hp < HPG; ++hp) {
    int h = g * HPG + hp;
    v8f acc[4] = {};
    for (int ks = 0; ks < 2; ++ks) {
      Frag a;
      {
        int s = sT * 16 + ln;
        const bf16_t* qr = qbf + ((size_t)s * H + h) * DK;
#pragma unroll
        for (int j = 0; j < 8; ++j) {
          int dd = ks * 32 + ((j < 4) ? 2 * j : 16 + 2 * (j - 4)) + (hi ? 8 : 0);
          a.u[j] = *(const unsigned int*)(qr + dd);
        }
      }
#pragma unroll
      for (int t = 0; t < 4; ++t) {
        Frag b;
        int c = t * 16 + ln;
        const bf16_t* kr = Kcb + ((size_t)(g * 64 + c)) * 64;
#pragma unroll
        for (int j = 0; j < 8; ++j) {
          int dd = ks * 32 + ((j < 4) ? 2 * j : 16 + 2 * (j - 4)) + (hi ? 8 : 0);
          b.u[j] = *(const unsigned int*)(kr + dd);
        }
        acc[t] = __builtin_amdgcn_wmma_f32_16x16x32_bf16(false, a.v, false, b.v,
                                                         (short)0, acc[t], false, false);
      }
    }
#pragma unroll
    for (int r = 0; r < 8; ++r) {
      int srow = sT * 16 + r + hi * 8;
      float mx = -1e30f;
#pragma unroll
      for (int t = 0; t < 4; ++t) {
        int c = t * 16 + ln;
        bool valid = (c < S_CMP) && (c * D_STR + (L_CMP - 1) <= srow);
        float sc = valid ? acc[t][r] * SCALE : -1e30f;
        acc[t][r] = sc;
        mx = fmaxf(mx, sc);
      }
#pragma unroll
      for (int off = 8; off > 0; off >>= 1) mx = fmaxf(mx, __shfl_xor(mx, off, 16));
      float sm = 0.f;
#pragma unroll
      for (int t = 0; t < 4; ++t) {
        float sc = acc[t][r];
        float pv = (sc > -1e29f) ? __expf(sc - mx) : 0.f;
        acc[t][r] = pv;
        sm += pv;
      }
#pragma unroll
      for (int off = 8; off > 0; off >>= 1) sm += __shfl_xor(sm, off, 16);
      float invs = (sm > 0.f) ? 1.f / sm : 0.f;
#pragma unroll
      for (int t = 0; t < 4; ++t) {
        acc[t][r] *= invs;
        lp[w][r + hi * 8][t * 16 + ln] = acc[t][r];
      }
    }
    __syncthreads();
#pragma unroll
    for (int q8 = 0; q8 < 8; ++q8) {
      int idx = lane + q8 * 32;
      int m = idx >> 4, blk = idx & 15;
      float s2 = 0.f;
      for (int c = 0; c < S_CMP; ++c) {
        int c0 = c * D_STR, m0 = blk * L_SEL;
        int lo = (c0 > m0) ? c0 : m0;
        int hi2 = ((c0 + L_CMP) < (m0 + L_SEL)) ? (c0 + L_CMP) : (m0 + L_SEL);
        if (hi2 > lo) s2 += lp[w][m][c] * ((float)(hi2 - lo) * (1.0f / L_CMP));
      }
      pacc[q8] += s2;
    }
    v8f oacc[4] = {};
    for (int ks = 0; ks < 2; ++ks) {
      Frag a;
#pragma unroll
      for (int j = 0; j < 8; ++j) {
        int c = ks * 32 + ((j < 4) ? 2 * j : 16 + 2 * (j - 4)) + (hi ? 8 : 0);
        a.u[j] = pk2(lp[w][ln][c], lp[w][ln][c + 1]);
      }
#pragma unroll
      for (int t = 0; t < 4; ++t) {
        Frag b;
        int d = t * 16 + ln;
#pragma unroll
        for (int j = 0; j < 8; ++j) {
          int c = ks * 32 + ((j < 4) ? 2 * j : 16 + 2 * (j - 4)) + (hi ? 8 : 0);
          b.u[j] = pk2b(Vcb[((size_t)(g * 64 + c)) * 64 + d],
                        Vcb[((size_t)(g * 64 + c + 1)) * 64 + d]);
        }
        oacc[t] = __builtin_amdgcn_wmma_f32_16x16x32_bf16(false, a.v, false, b.v,
                                                          (short)0, oacc[t], false, false);
      }
    }
#pragma unroll
    for (int t = 0; t < 4; ++t) {
      int d = t * 16 + ln;
#pragma unroll
      for (int r = 0; r < 8; ++r) {
        int s = sT * 16 + r + hi * 8;
        Ocmp[((size_t)s * H + h) * DV + d] = oacc[t][r];
      }
    }
    __syncthreads();
  }
#pragma unroll
  for (int q8 = 0; q8 < 8; ++q8) {
    int idx = lane + q8 * 32;
    int m = idx >> 4, blk = idx & 15;
    int s = sT * 16 + m;
    pslc[((size_t)s * G + g) * N_BLK + blk] = pacc[q8];
  }
}

// ---------------------------------------------------------------- top-k
__global__ __launch_bounds__(256) void k_topk(const float* __restrict__ pslc,
                                              int* __restrict__ topidx) {
  int i = blockIdx.x * 256 + threadIdx.x;
  if (i >= S * G) return;
  int s = i >> 2;
  float lg[N_BLK];
  for (int m = 0; m < N_BLK; ++m) {
    bool causal = (m * L_SEL) <= s;
    bool forced = (m == 0) || (m == (s >> 6));
    lg[m] = causal ? (pslc[(size_t)i * N_BLK + m] + (forced ? 1e6f : 0.f)) : -1e9f;
  }
  unsigned used = 0;
  for (int k = 0; k < N_SEL; ++k) {
    int best = 0;
    float bv = -1e38f;
    for (int m = 0; m < N_BLK; ++m)
      if (!((used >> m) & 1u) && lg[m] > bv) { bv = lg[m]; best = m; }
    used |= 1u << best;
    topidx[(size_t)i * N_SEL + k] = best;
  }
}

// ------------------------------------------------------ selected attention
__global__ __launch_bounds__(128) void k_sel_attn(const bf16_t* __restrict__ qbf,
                                                  const bf16_t* __restrict__ kbf,
                                                  const bf16_t* __restrict__ vbf,
                                                  const int* __restrict__ topidx,
                                                  float* __restrict__ Osel) {
  __shared__ float sp[4][4][512];
  __shared__ int bsel[4][8];
  int w = threadIdx.x >> 5, lane = threadIdx.x & 31;
  int hi = lane >> 4, ln = lane & 15;
  int wid = blockIdx.x * 4 + w;
  int g = wid & (G - 1), s = wid >> 2;
  if (lane < 8) bsel[w][lane] = topidx[((size_t)s * G + g) * N_SEL + lane];
  __syncthreads();

  Frag aS[2];
  for (int ks = 0; ks < 2; ++ks) {
    const bf16_t* qr = qbf + ((size_t)s * H + g * HPG + (ln & 3)) * DK;
#pragma unroll
    for (int j = 0; j < 8; ++j) {
      int dd = ks * 32 + ((j < 4) ? 2 * j : 16 + 2 * (j - 4)) + (hi ? 8 : 0);
      aS[ks].u[j] = (ln < HPG) ? *(const unsigned int*)(qr + dd) : 0u;
    }
  }
  for (int tt = 0; tt < 32; ++tt) {
    int tokBase = bsel[w][tt >> 2] * L_SEL + (tt & 3) * 16;
    v8f acc = {};
#pragma unroll
    for (int ks = 0; ks < 2; ++ks) {
      Frag b;
      int tok = tokBase + ln;
      const bf16_t* kr = kbf + (size_t)tok * KV + g * DK;
#pragma unroll
      for (int j = 0; j < 8; ++j) {
        int dd = ks * 32 + ((j < 4) ? 2 * j : 16 + 2 * (j - 4)) + (hi ? 8 : 0);
        b.u[j] = *(const unsigned int*)(kr + dd);
      }
      acc = __builtin_amdgcn_wmma_f32_16x16x32_bf16(false, aS[ks].v, false, b.v,
                                                    (short)0, acc, false, false);
    }
    if (hi == 0) {
      int tok = tokBase + ln;
#pragma unroll
      for (int r = 0; r < HPG; ++r)
        sp[w][r][tt * 16 + ln] = (tok <= s) ? acc[r] * SCALE : -1e30f;
    }
  }
  __syncthreads();
#pragma unroll
  for (int h = 0; h < HPG; ++h) {
    float mx = -1e30f;
    for (int i = lane; i < 512; i += 32) mx = fmaxf(mx, sp[w][h][i]);
#pragma unroll
    for (int off = 16; off > 0; off >>= 1) mx = fmaxf(mx, __shfl_xor(mx, off, 32));
    float sm = 0.f;
    for (int i = lane; i < 512; i += 32) {
      float v = sp[w][h][i];
      float p = (v > -1e29f) ? __expf(v - mx) : 0.f;
      sp[w][h][i] = p;
      sm += p;
    }
#pragma unroll
    for (int off = 16; off > 0; off >>= 1) sm += __shfl_xor(sm, off, 32);
    float invs = (sm > 0.f) ? 1.f / sm : 0.f;
    for (int i = lane; i < 512; i += 32) sp[w][h][i] *= invs;
  }
  __syncthreads();
  v8f oacc[4] = {};
  for (int ks = 0; ks < 16; ++ks) {
    Frag a;
#pragma unroll
    for (int j = 0; j < 8; ++j) {
      int kk = ks * 32 + ((j < 4) ? 2 * j : 16 + 2 * (j - 4)) + (hi ? 8 : 0);
      float p0 = (ln < HPG) ? sp[w][ln & 3][kk] : 0.f;
      float p1 = (ln < HPG) ? sp[w][ln & 3][kk + 1] : 0.f;
      a.u[j] = pk2(p0, p1);
    }
#pragma unroll
    for (int t = 0; t < 4; ++t) {
      Frag b;
      int d = t * 16 + ln;
#pragma unroll
      for (int j = 0; j < 8; ++j) {
        int kk = ks * 32 + ((j < 4) ? 2 * j : 16 + 2 * (j - 4)) + (hi ? 8 : 0);
        int tok = bsel[w][kk >> 6] * L_SEL + (kk & 63);
        b.u[j] = pk2b(vbf[(size_t)tok * KV + g * DK + d],
                      vbf[(size_t)(tok + 1) * KV + g * DK + d]);
      }
      oacc[t] = __builtin_amdgcn_wmma_f32_16x16x32_bf16(false, a.v, false, b.v,
                                                        (short)0, oacc[t], false, false);
    }
  }
  if (hi == 0) {
#pragma unroll
    for (int t = 0; t < 4; ++t) {
      int d = t * 16 + ln;
#pragma unroll
      for (int r = 0; r < HPG; ++r)
        Osel[((size_t)s * H + g * HPG + r) * DV + d] = oacc[t][r];
    }
  }
}

// ------------------------------- window attention (16-query x 1-head tiles)
// One wave per (16-query tile, head). Union window of the 16 queries is
// [s0-511, s0+15] -> 544 padded token slots (34 tiles); per-row mask m<=i<=511+m.
constexpr int WT = 34;

__global__ __launch_bounds__(32) void k_win_attn(const bf16_t* __restrict__ qbf,
                                                 const bf16_t* __restrict__ kbf,
                                                 const bf16_t* __restrict__ vbf,
                                                 float* __restrict__ Owin) {
  __shared__ float sp[16][WT * 16];
  int lane = threadIdx.x & 31;
  int hi = lane >> 4, ln = lane & 15;
  int wid = blockIdx.x;
  int h = wid & (H - 1), sT = wid >> 4;
  int g = h >> 2;
  int base = sT * 16 - (W_WIN - 1);

  Frag aS[2];
  for (int ks = 0; ks < 2; ++ks) {
    int s = sT * 16 + ln;
    const bf16_t* qr = qbf + ((size_t)s * H + h) * DK;
#pragma unroll
    for (int j = 0; j < 8; ++j) {
      int dd = ks * 32 + ((j < 4) ? 2 * j : 16 + 2 * (j - 4)) + (hi ? 8 : 0);
      aS[ks].u[j] = *(const unsigned int*)(qr + dd);
    }
  }
  for (int tt = 0; tt < WT; ++tt) {
    int i0 = tt * 16 + ln;
    int tok = base + i0;
    int tokc = tok < 0 ? 0 : (tok > S - 1 ? S - 1 : tok);
    v8f acc = {};
#pragma unroll
    for (int ks = 0; ks < 2; ++ks) {
      Frag b;
      const bf16_t* kr = kbf + (size_t)tokc * KV + g * DK;
#pragma unroll
      for (int j = 0; j < 8; ++j) {
        int dd = ks * 32 + ((j < 4) ? 2 * j : 16 + 2 * (j - 4)) + (hi ? 8 : 0);
        b.u[j] = *(const unsigned int*)(kr + dd);
      }
      acc = __builtin_amdgcn_wmma_f32_16x16x32_bf16(false, aS[ks].v, false, b.v,
                                                    (short)0, acc, false, false);
    }
#pragma unroll
    for (int r = 0; r < 8; ++r) {
      int m = r + hi * 8;
      bool valid = (i0 >= m) && (i0 <= 511 + m) && (tok >= 0);
      sp[m][i0] = valid ? acc[r] * SCALE : -1e30f;
    }
  }
  __syncthreads();
  for (int m = 0; m < 16; ++m) {
    float mx = -1e30f;
    for (int i = lane; i < WT * 16; i += 32) mx = fmaxf(mx, sp[m][i]);
#pragma unroll
    for (int off = 16; off > 0; off >>= 1) mx = fmaxf(mx, __shfl_xor(mx, off, 32));
    float sm = 0.f;
    for (int i = lane; i < WT * 16; i += 32) {
      float v = sp[m][i];
      float p = (v > -1e29f) ? __expf(v - mx) : 0.f;
      sp[m][i] = p;
      sm += p;
    }
#pragma unroll
    for (int off = 16; off > 0; off >>= 1) sm += __shfl_xor(sm, off, 32);
    float invs = (sm > 0.f) ? 1.f / sm : 0.f;
    for (int i = lane; i < WT * 16; i += 32) sp[m][i] *= invs;
  }
  __syncthreads();
  v8f oacc[4] = {};
  for (int ks = 0; ks < WT / 2; ++ks) {
    Frag a;
#pragma unroll
    for (int j = 0; j < 8; ++j) {
      int kk = ks * 32 + ((j < 4) ? 2 * j : 16 + 2 * (j - 4)) + (hi ? 8 : 0);
      a.u[j] = pk2(sp[ln][kk], sp[ln][kk + 1]);
    }
#pragma unroll
    for (int t = 0; t < 4; ++t) {
      Frag b;
      int d = t * 16 + ln;
#pragma unroll
      for (int j = 0; j < 8; ++j) {
        int kk = ks * 32 + ((j < 4) ? 2 * j : 16 + 2 * (j - 4)) + (hi ? 8 : 0);
        int t0 = base + kk;
        int t1 = t0 + 1;
        t0 = t0 < 0 ? 0 : (t0 > S - 1 ? S - 1 : t0);
        t1 = t1 < 0 ? 0 : (t1 > S - 1 ? S - 1 : t1);
        b.u[j] = pk2b(vbf[(size_t)t0 * KV + g * DK + d],
                      vbf[(size_t)t1 * KV + g * DK + d]);
      }
      oacc[t] = __builtin_amdgcn_wmma_f32_16x16x32_bf16(false, a.v, false, b.v,
                                                        (short)0, oacc[t], false, false);
    }
  }
#pragma unroll
  for (int t = 0; t < 4; ++t) {
    int d = t * 16 + ln;
#pragma unroll
    for (int r = 0; r < 8; ++r) {
      int s = sT * 16 + r + hi * 8;
      Owin[((size_t)s * H + h) * DV + d] = oacc[t][r];
    }
  }
}

// ---------------------------------------------------------------- gate MLP
__global__ __launch_bounds__(256) void k_gate(const float* __restrict__ qf,
                                              const float* __restrict__ g1w,
                                              const float* __restrict__ g1b,
                                              const float* __restrict__ g2w,
                                              const float* __restrict__ g2b,
                                              float* __restrict__ pg) {
  int i = blockIdx.x * 256 + threadIdx.x;
  if (i >= S * G) return;
  int g = i & (G - 1), s = i >> 2;
  float qg[DK];
  for (int d = 0; d < DK; ++d) {
    float a = 0.f;
    for (int hp = 0; hp < HPG; ++hp) a += qf[((size_t)s * H + g * HPG + hp) * DK + d];
    qg[d] = a * 0.25f;
  }
  float h1[GATE_H];
  for (int j = 0; j < GATE_H; ++j) {
    float a = g1b[j];
    for (int d = 0; d < DK; ++d) a += qg[d] * g1w[j * DK + d];
    h1[j] = a / (1.f + __expf(-a));
  }
  float gl[3];
  for (int o = 0; o < 3; ++o) {
    float a = g2b[o];
    for (int j = 0; j < GATE_H; ++j) a += h1[j] * g2w[o * GATE_H + j];
    gl[o] = a; // TAU == 1
  }
  int amax = 0;
  if (gl[1] > gl[amax]) amax = 1;
  if (gl[2] > gl[amax]) amax = 2;
  float m1 = gl[amax];
  float m2 = -1e38f;
  for (int o = 0; o < 3; ++o)
    if (o != amax) m2 = fmaxf(m2, gl[o]);
  float e0 = __expf(gl[0] - m1), e1 = __expf(gl[1] - m1), e2 = __expf(gl[2] - m1);
  float inv = 1.f / (e0 + e1 + e2);
  float p0 = e0 * inv, p1 = e1 * inv, p2 = e2 * inv;
  if (m1 - m2 > 50.f) {
    p0 = (amax == 0) ? 1.f : 0.f;
    p1 = (amax == 1) ? 1.f : 0.f;
    p2 = (amax == 2) ? 1.f : 0.f;
  }
  pg[(size_t)i * 3 + 0] = p0;
  pg[(size_t)i * 3 + 1] = p1;
  pg[(size_t)i * 3 + 2] = p2;
}

// ---------------------------------------------------------------- combine
__global__ __launch_bounds__(256) void k_combine(const float* __restrict__ Ocmp,
                                                 const float* __restrict__ Osel,
                                                 const float* __restrict__ Owin,
                                                 const float* __restrict__ pg,
                                                 bf16_t* __restrict__ Obf) {
  int i = blockIdx.x * 256 + threadIdx.x;
  if (i >= S * H * DV) return;
  int h = (i >> 6) & (H - 1), s = i >> 10;
  int g = h >> 2;
  const float* pgp = pg + ((size_t)s * G + g) * 3;
  float o = pgp[0] * Ocmp[i] + pgp[1] * Osel[i] + pgp[2] * Owin[i];
  Obf[i] = (bf16_t)o;
}

// =================================================================== launch
extern "C" void kernel_launch(void* const* d_in, const int* in_sizes, int n_in,
                              void* d_out, int out_size, void* d_ws, size_t ws_size,
                              hipStream_t stream) {
  (void)in_sizes; (void)n_in; (void)out_size; (void)ws_size;
  const float* x      = (const float*)d_in[0];
  const float* wq     = (const float*)d_in[1];
  const float* wk_sel = (const float*)d_in[2];
  const float* wv_sel = (const float*)d_in[3];
  const float* wk_win = (const float*)d_in[4];
  const float* wv_win = (const float*)d_in[5];
  const float* wk_cmp = (const float*)d_in[6];
  const float* wv_cmp = (const float*)d_in[7];
  const float* w_out  = (const float*)d_in[8];
  const float* g1w    = (const float*)d_in[9];
  const float* g1b    = (const float*)d_in[10];
  const float* g2w    = (const float*)d_in[11];
  const float* g2b    = (const float*)d_in[12];

  size_t off = 0;
  auto carve = [&](size_t bytes) -> void* {
    void* r = (char*)d_ws + off;
    off = (off + bytes + 255) & ~(size_t)255;
    return r;
  };
  bf16_t* xbf  = (bf16_t*)carve((size_t)S * DIM * 2);
  bf16_t* wqbf = (bf16_t*)carve((size_t)DIM * DIM * 2);
  bf16_t* wksb = (bf16_t*)carve((size_t)KV * DIM * 2);
  bf16_t* wvsb = (bf16_t*)carve((size_t)KV * DIM * 2);
  bf16_t* wkwb = (bf16_t*)carve((size_t)KV * DIM * 2);
  bf16_t* wvwb = (bf16_t*)carve((size_t)KV * DIM * 2);
  bf16_t* wkcb = (bf16_t*)carve((size_t)KV * DIM * 2);
  bf16_t* wvcb = (bf16_t*)carve((size_t)KV * DIM * 2);
  bf16_t* wob  = (bf16_t*)carve((size_t)DIM * DIM * 2);
  float*  qf   = (float*) carve((size_t)S * H * DK * 4);
  bf16_t* qbf  = (bf16_t*)carve((size_t)S * H * DK * 2);
  float*  ksf  = (float*) carve((size_t)S * KV * 4);
  float*  kwf  = (float*) carve((size_t)S * KV * 4);
  float*  kcf  = (float*) carve((size_t)S * KV * 4);
  float*  vsf  = (float*) carve((size_t)S * KV * 4);
  float*  vwf  = (float*) carve((size_t)S * KV * 4);
  float*  vcf  = (float*) carve((size_t)S * KV * 4);
  bf16_t* ksb  = (bf16_t*)carve((size_t)S * KV * 2);
  bf16_t* kwb  = (bf16_t*)carve((size_t)S * KV * 2);
  bf16_t* kcb  = (bf16_t*)carve((size_t)S * KV * 2);
  bf16_t* vsb  = (bf16_t*)carve((size_t)S * KV * 2);
  bf16_t* vwb  = (bf16_t*)carve((size_t)S * KV * 2);
  bf16_t* Kcb  = (bf16_t*)carve((size_t)G * 64 * 64 * 2);
  bf16_t* Vcb  = (bf16_t*)carve((size_t)G * 64 * 64 * 2);
  float*  pslc = (float*) carve((size_t)S * G * N_BLK * 4);
  int*    tidx = (int*)   carve((size_t)S * G * N_SEL * 4);
  float*  Ocmp = (float*) carve((size_t)S * H * DV * 4);
  float*  Osel = (float*) carve((size_t)S * H * DV * 4);
  float*  Owin = (float*) carve((size_t)S * H * DV * 4);
  float*  pgb  = (float*) carve((size_t)S * G * 3 * 4);
  bf16_t* Obf  = (bf16_t*)carve((size_t)S * H * DV * 2);

  auto cvt = [&](const float* srcp, bf16_t* dstp, int n) {
    k_cvt_bf16<<<(n + 255) / 256, 256, 0, stream>>>(srcp, dstp, n);
  };
  cvt(x, xbf, S * DIM);
  cvt(wq, wqbf, DIM * DIM);
  cvt(wk_sel, wksb, KV * DIM);
  cvt(wv_sel, wvsb, KV * DIM);
  cvt(wk_win, wkwb, KV * DIM);
  cvt(wv_win, wvwb, KV * DIM);
  cvt(wk_cmp, wkcb, KV * DIM);
  cvt(wv_cmp, wvcb, KV * DIM);
  cvt(w_out, wob, DIM * DIM);

  auto gemm = [&](const bf16_t* A, const bf16_t* W, float* C, int M, int N, int K) {
    int blocks = (M / BM) * (N / BN);
    k_gemm_bf16<<<blocks, 256, 0, stream>>>(A, W, C, M, N, K);
  };
  gemm(xbf, wqbf, qf, S, H * DK, DIM);
  gemm(xbf, wksb, ksf, S, KV, DIM);
  gemm(xbf, wvsb, vsf, S, KV, DIM);
  gemm(xbf, wkwb, kwf, S, KV, DIM);
  gemm(xbf, wvwb, vwf, S, KV, DIM);
  gemm(xbf, wkcb, kcf, S, KV, DIM);
  gemm(xbf, wvcb, vcf, S, KV, DIM);

  k_rope_q<<<(S * H * 32 + 255) / 256, 256, 0, stream>>>(qf, qbf);
  k_rope_k<<<(S * G * 32 + 255) / 256, 256, 0, stream>>>(ksf, ksb);
  k_rope_k<<<(S * G * 32 + 255) / 256, 256, 0, stream>>>(kwf, kwb);
  k_rope_k<<<(S * G * 32 + 255) / 256, 256, 0, stream>>>(kcf, kcb);
  cvt(vsf, vsb, S * KV);
  cvt(vwf, vwb, S * KV);

  k_pool<<<(G * 64 * 64 + 255) / 256, 256, 0, stream>>>(kcf, vcf, Kcb, Vcb);
  k_cmp_attn<<<(S / 16) * G / 4, 128, 0, stream>>>(qbf, Kcb, Vcb, Ocmp, pslc);
  k_topk<<<(S * G + 255) / 256, 256, 0, stream>>>(pslc, tidx);
  k_sel_attn<<<S * G / 4, 128, 0, stream>>>(qbf, ksb, vsb, tidx, Osel);
  k_win_attn<<<(S / 16) * H, 32, 0, stream>>>(qbf, kwb, vwb, Owin);
  k_gate<<<(S * G + 255) / 256, 256, 0, stream>>>(qf, g1w, g1b, g2w, g2b, pgb);
  k_combine<<<(S * H * DV + 255) / 256, 256, 0, stream>>>(Ocmp, Osel, Owin, pgb, Obf);
  gemm(Obf, wob, (float*)d_out, S, DIM, H * DV);
}
